// EGnnNetwork_53523882442973
// MI455X (gfx1250) — matrix operand order repaired and checked by reading, not compiled, not run on hardware
//
#include <hip/hip_runtime.h>
#include <hip/hip_bf16.h>
#include <math.h>

#define BB 2
#define NN 2048
#define KK 48

typedef __attribute__((ext_vector_type(16))) _Float16 v16h;
typedef __attribute__((ext_vector_type(8)))  float    v8f;

#define FRAG 512  // halves per 16x32 (A) / 32x16 (B) fragment: 32 lanes x 16 halves

// layer tiling (K padded to mult of 32, N padded to mult of 16)
#define E1_KT 7    // Kin 193 -> 224
#define E1_NT 25   // Nout 386 -> 400
#define E2_KT 13   // Kin 386 -> 416
#define E2_NT 4    // Nout 64
#define H1_KT 2    // Kin 64
#define H1_NT 16   // Nout 256
#define H2_KT 8    // Kin 256
#define H2_NT 4    // Nout 64

// workspace offsets (halves)
#define OFF_E1 0
#define SZ_E1 (E1_KT*E1_NT*FRAG)
#define OFF_E2 (OFF_E1 + SZ_E1)
#define SZ_E2 (E2_KT*E2_NT*FRAG)
#define OFF_H1 (OFF_E2 + SZ_E2)
#define SZ_H1W (H1_KT*H1_NT*FRAG)
#define OFF_H2 (OFF_H1 + SZ_H1W)
#define SZ_H2W (H2_KT*H2_NT*FRAG)

// dynamic shared layout (bytes)
#define SM_REL   0                        // 48*64*3 f32  = 36864
#define SM_NORM  (SM_REL  + 36864)        // 48*64 f32    = 12288
#define SM_MIJ   (SM_NORM + 12288)        // 48*64 f32    = 12288
#define SM_HTW   (SM_MIJ  + 12288)        // 48*64 f32    = 12288
#define SM_L1    (SM_HTW  + 12288)        // 3*13 frags f16 = 39936
#define SM_XH    (SM_L1   + 39936)        // union: X (3*7) / H1 (3*8) frags = 24576
#define SM_M     (SM_XH   + 24576)        // 3*2 frags f16 = 6144
#define SM_MISC  (SM_M    + 6144)         // misc ~3456
#define SM_TOTAL (SM_MISC + 3456)
#define ZERO_BYTES (39936 + 24576 + 6144) // sL1 + sXH + sM contiguous

__device__ __forceinline__ float silu_f(float y) {
  return y / (1.f + __expf(-y));
}

// Write value into A-fragment layout: fragment (mt,kt), element (m in 0..15, k in 0..31)
// lane = m + 16*g ; g=(k>>3)&1 ; h=(k&7) + (k>=16 ? 8 : 0)
__device__ __forceinline__ void writeA(_Float16* base, int KT, int e, int f, float val) {
  int mt = e >> 4, m = e & 15;
  int kt = f >> 5, kk = f & 31;
  int g = (kk >> 3) & 1;
  int h = (kk & 7) + ((kk & 16) ? 8 : 0);
  base[((mt * KT + kt) << 9) + ((m + (g << 4)) << 4) + h] = (_Float16)val;
}

__device__ __forceinline__ v8f wmma_f16(v16h a, v16h b, v8f c) {
  return __builtin_amdgcn_wmma_f32_16x16x32_f16(false, a, false, b, (short)0, c, false, false);
}

// Re-layout f32 [Kin][Nout] weights into f16 B-fragment-major layout:
// frag = kt*NT + nt ; within frag: lane*16 + h ; K = kt*32 + (lane/16)*8 + (h%8) + (h/8)*16 ; N = nt*16 + lane%16
__global__ void prep_weights(const float* __restrict__ W, _Float16* __restrict__ dst,
                             int Kin, int Nout, int NT, int total) {
  int e = blockIdx.x * blockDim.x + threadIdx.x;
  if (e >= total) return;
  int frag = e >> 9;
  int w = e & 511;
  int lane = w >> 4;
  int h = w & 15;
  int kt = frag / NT, nt = frag % NT;
  int k = kt * 32 + ((lane >> 4) << 3) + (h & 7) + ((h >> 3) << 4);
  int n = nt * 16 + (lane & 15);
  float v = 0.f;
  if (k < Kin && n < Nout) v = W[(size_t)k * Nout + n];
  dst[e] = (_Float16)v;
}

__global__ __launch_bounds__(256)
void egnn_main(const float* __restrict__ f0, const float* __restrict__ f1,
               const float* __restrict__ reldist, const int* __restrict__ nidx,
               const unsigned char* __restrict__ nmask,
               const float* __restrict__ ln_g, const float* __restrict__ ln_b,
               const float* __restrict__ eb1, const float* __restrict__ eb2,
               const float* __restrict__ hb1, const float* __restrict__ hb2,
               const float* __restrict__ nW1, const float* __restrict__ nb1,
               const float* __restrict__ nW2, const float* __restrict__ nb2,
               const float* __restrict__ hn_scale, const float* __restrict__ hn_bias,
               const float* __restrict__ gW, const float* __restrict__ gb,
               const _Float16* __restrict__ wfrag, float* __restrict__ out) {
  extern __shared__ float4 smem4[];
  char* smem = (char*)smem4;
  float*    s_rel  = (float*)(smem + SM_REL);     // [48][64][3]
  float*    s_norm = (float*)(smem + SM_NORM);    // [48][64]
  float*    s_mij  = (float*)(smem + SM_MIJ);     // [48][64]
  float*    s_htw  = (float*)(smem + SM_HTW);     // [48][64]
  _Float16* sL1    = (_Float16*)(smem + SM_L1);   // E1 out, A-frags for E2
  _Float16* sX     = (_Float16*)(smem + SM_XH);   // edge input A-frags (dead after E1)
  _Float16* sH1    = (_Float16*)(smem + SM_XH);   // H1 out A-frags (aliases sX)
  _Float16* sM     = (_Float16*)(smem + SM_M);    // m_ij A-frags for H1
  float* s_nodes = (float*)(smem + SM_MISC);      // 64
  float* s_hti   = s_nodes + 64;                  // 192
  float* s_reld  = s_hti + 192;                   // 48
  float* s_mi    = s_reld + 48;                   // 64
  float* s_nm    = s_mi + 64;                     // 128
  float* s_h1n   = s_nm + 128;                    // 128
  float* s_no    = s_h1n + 128;                   // 64
  float* s_gate  = s_no + 64;                     // 64
  float* s_stats = s_gate + 64;                   // 2
  int*   s_jidx  = (int*)(s_stats + 2);           // 48
  int*   s_maski = s_jidx + 48;                   // 48

  const int tid  = threadIdx.x;
  const int lane = tid & 31;
  const int wave = tid >> 5;
  const int bn   = blockIdx.x;       // b*N + n
  const int b    = bn / NN;

  // ---- phase A: zero fragment regions, load node-i data ----
  {
    int* z = (int*)(smem + SM_L1);
    for (int i = tid; i < ZERO_BYTES / 4; i += 256) z[i] = 0;
  }
  if (tid < 64)  s_nodes[tid] = f0[(size_t)bn * 64 + tid];
  if (tid < 192) s_hti[tid]   = f1[(size_t)bn * 192 + tid];
  if (tid < KK) {
    s_jidx[tid]  = nidx[(size_t)bn * KK + tid];
    s_reld[tid]  = reldist[(size_t)bn * KK + tid];
    s_maski[tid] = (int)nmask[(size_t)bn * KK + tid];
  }
  __syncthreads();

  // LayerNorm stats (tiny; one lane)
  if (tid == 0) {
    float mu = 0.f;
    for (int d = 0; d < 64; ++d) mu += s_nodes[d];
    mu *= (1.f / 64.f);
    float var = 0.f;
    for (int d = 0; d < 64; ++d) { float t = s_nodes[d] - mu; var += t * t; }
    var *= (1.f / 64.f);
    s_stats[0] = mu;
    s_stats[1] = rsqrtf(var + 1e-5f);
  }

  // ---- phase B: gather neighbors, build X fragments + rel geometry ----
  for (int it = tid; it < KK * 64; it += 256) {
    int e = it >> 6, d = it & 63;
    size_t jb = (size_t)b * NN + (size_t)s_jidx[e];
    float nj = f0[jb * 64 + d];
    const float* hj = f1 + jb * 192 + d * 3;
    float r0 = s_hti[d * 3 + 0] - hj[0];
    float r1 = s_hti[d * 3 + 1] - hj[1];
    float r2 = s_hti[d * 3 + 2] - hj[2];
    float nrm = sqrtf(r0 * r0 + r1 * r1 + r2 * r2);
    float* rp = s_rel + (size_t)(e * 64 + d) * 3;
    rp[0] = r0; rp[1] = r1; rp[2] = r2;
    s_norm[e * 64 + d] = nrm;
    writeA(sX, E1_KT, e, d,        s_nodes[d]);
    writeA(sX, E1_KT, e, 64 + d,   nj);
    writeA(sX, E1_KT, e, 128 + d,  nrm);
    if (d == 0) writeA(sX, E1_KT, e, 192, s_reld[e]);
  }
  __syncthreads();

  // ---- E1: X(48x224) @ eW1 -> 48x386, SiLU, A-frags for E2 ----
  // nt-outer / kt-mid / mt-inner: each B fragment loaded once per block,
  // reused across all 3 M-tiles.
  for (int nt = wave; nt < E1_NT; nt += 8) {
    v8f acc[3] = {{}, {}, {}};
    for (int kt = 0; kt < E1_KT; ++kt) {
      v16h bm = *(const v16h*)(wfrag + OFF_E1 + ((kt * E1_NT + nt) << 9) + (lane << 4));
#pragma unroll
      for (int mt = 0; mt < 3; ++mt) {
        v16h a = *(const v16h*)(sX + ((mt * E1_KT + kt) << 9) + (lane << 4));
        acc[mt] = wmma_f16(a, bm, acc[mt]);
      }
    }
    int nb = nt * 16 + (lane & 15);
    if (nb < 386) {
      float bias = eb1[nb];
      int ktn = nt >> 1;
      int kk = ((nt & 1) << 4) + (lane & 15);
      int g = (kk >> 3) & 1;
      int h = (kk & 7) + ((kk & 16) ? 8 : 0);
#pragma unroll
      for (int mt = 0; mt < 3; ++mt) {
        for (int v = 0; v < 8; ++v) {
          float y = silu_f(acc[mt][v] + bias);
          int m = v + ((lane >> 4) << 3);
          sL1[((mt * E2_KT + ktn) << 9) + ((m + (g << 4)) << 4) + h] = (_Float16)y;
        }
      }
    }
  }
  __syncthreads();

  // ---- E2: 48x416 @ eW2 -> m_ij (48x64), SiLU ----
  for (int nt = wave; nt < E2_NT; nt += 8) {
    v8f acc[3] = {{}, {}, {}};
    for (int kt = 0; kt < E2_KT; ++kt) {
      v16h bm = *(const v16h*)(wfrag + OFF_E2 + ((kt * E2_NT + nt) << 9) + (lane << 4));
#pragma unroll
      for (int mt = 0; mt < 3; ++mt) {
        v16h a = *(const v16h*)(sL1 + ((mt * E2_KT + kt) << 9) + (lane << 4));
        acc[mt] = wmma_f16(a, bm, acc[mt]);
      }
    }
    int nb = nt * 16 + (lane & 15);
    float bias = eb2[nb];
    int ktn = nt >> 1;
    int kk = ((nt & 1) << 4) + (lane & 15);
    int g = (kk >> 3) & 1;
    int h = (kk & 7) + ((kk & 16) ? 8 : 0);
#pragma unroll
    for (int mt = 0; mt < 3; ++mt) {
      for (int v = 0; v < 8; ++v) {
        float y = silu_f(acc[mt][v] + bias);
        int m = v + ((lane >> 4) << 3);
        int row = mt * 16 + m;
        s_mij[row * 64 + nb] = y;
        sM[((mt * H1_KT + ktn) << 9) + ((m + (g << 4)) << 4) + h] = (_Float16)y;
      }
    }
  }
  __syncthreads();

  // ---- H1: m_ij(48x64) @ hW1 -> 48x256, SiLU (overwrites dead X region) ----
  for (int nt = wave; nt < H1_NT; nt += 8) {
    v8f acc[3] = {{}, {}, {}};
    for (int kt = 0; kt < H1_KT; ++kt) {
      v16h bm = *(const v16h*)(wfrag + OFF_H1 + ((kt * H1_NT + nt) << 9) + (lane << 4));
#pragma unroll
      for (int mt = 0; mt < 3; ++mt) {
        v16h a = *(const v16h*)(sM + ((mt * H1_KT + kt) << 9) + (lane << 4));
        acc[mt] = wmma_f16(a, bm, acc[mt]);
      }
    }
    int nb = nt * 16 + (lane & 15);
    float bias = hb1[nb];
    int ktn = nt >> 1;
    int kk = ((nt & 1) << 4) + (lane & 15);
    int g = (kk >> 3) & 1;
    int h = (kk & 7) + ((kk & 16) ? 8 : 0);
#pragma unroll
    for (int mt = 0; mt < 3; ++mt) {
      for (int v = 0; v < 8; ++v) {
        float y = silu_f(acc[mt][v] + bias);
        int m = v + ((lane >> 4) << 3);
        sH1[((mt * H2_KT + ktn) << 9) + ((m + (g << 4)) << 4) + h] = (_Float16)y;
      }
    }
  }
  __syncthreads();

  // ---- H2: 48x256 @ hW2 -> htw (48x64), no activation ----
  for (int nt = wave; nt < H2_NT; nt += 8) {
    v8f acc[3] = {{}, {}, {}};
    for (int kt = 0; kt < H2_KT; ++kt) {
      v16h bm = *(const v16h*)(wfrag + OFF_H2 + ((kt * H2_NT + nt) << 9) + (lane << 4));
#pragma unroll
      for (int mt = 0; mt < 3; ++mt) {
        v16h a = *(const v16h*)(sH1 + ((mt * H2_KT + kt) << 9) + (lane << 4));
        acc[mt] = wmma_f16(a, bm, acc[mt]);
      }
    }
    int nb = nt * 16 + (lane & 15);
    float bias = hb2[nb];
#pragma unroll
    for (int mt = 0; mt < 3; ++mt) {
      for (int v = 0; v < 8; ++v) {
        int m = v + ((lane >> 4) << 3);
        int row = mt * 16 + m;
        s_htw[row * 64 + nb] = acc[mt][v] + bias;
      }
    }
  }
  __syncthreads();

  // ---- node epilogue (VALU, tiny) ----
  if (tid < 64) {
    float s = 0.f;
    for (int e = 0; e < KK; ++e)
      if (s_maski[e]) s += s_mij[e * 64 + tid];
    s_mi[tid] = s;
    float mu = s_stats[0], inv = s_stats[1];
    s_nm[tid]      = (s_nodes[tid] - mu) * inv * ln_g[tid] + ln_b[tid];
    s_nm[64 + tid] = s;
  }
  __syncthreads();
  if (tid < 128) {
    float acc = nb1[tid];
    for (int k = 0; k < 128; ++k) acc += s_nm[k] * nW1[k * 128 + tid];
    s_h1n[tid] = silu_f(acc);
  }
  __syncthreads();
  if (tid < 64) {
    float acc = nb2[tid];
    for (int j = 0; j < 128; ++j) acc += s_h1n[j] * nW2[j * 64 + tid];
    s_no[tid] = acc + s_nodes[tid];
  }
  __syncthreads();
  if (tid < 64) {
    float acc = gb[tid];
    for (int e = 0; e < 64; ++e) acc += s_no[e] * gW[e * 64 + tid];
    s_gate[tid] = 1.f / (1.f + __expf(-acc));
  }
  __syncthreads();
  if (tid < 192) {
    int d = tid / 3, c = tid - d * 3;
    float sc = hn_scale[d], bi = hn_bias[d];
    float acc = 0.f;
    for (int e = 0; e < KK; ++e) {
      float nrm = s_norm[e * 64 + d];
      float fac = (nrm * sc + bi) / fmaxf(nrm, 1e-8f);
      acc += s_rel[(e * 64 + d) * 3 + c] * fac * s_htw[e * 64 + d];
    }
    float f1v = (s_hti[d * 3 + c] + acc) * s_gate[d];
    out[((size_t)bn * 64 + d) * 4 + 1 + c] = f1v;
  }
  if (tid < 64) out[((size_t)bn * 64 + tid) * 4] = s_no[tid];
}

extern "C" void kernel_launch(void* const* d_in, const int* in_sizes, int n_in,
                              void* d_out, int out_size, void* d_ws, size_t ws_size,
                              hipStream_t stream) {
  (void)in_sizes; (void)n_in; (void)out_size; (void)ws_size;
  const float* f0       = (const float*)d_in[0];
  const float* f1       = (const float*)d_in[1];
  const float* reldist  = (const float*)d_in[2];
  const int*   nidx     = (const int*)d_in[3];
  const unsigned char* nmask = (const unsigned char*)d_in[4];
  const float* ln_g = (const float*)d_in[5];
  const float* ln_b = (const float*)d_in[6];
  const float* eW1  = (const float*)d_in[7];
  const float* eb1  = (const float*)d_in[8];
  const float* eW2  = (const float*)d_in[9];
  const float* eb2  = (const float*)d_in[10];
  const float* hW1  = (const float*)d_in[11];
  const float* hb1  = (const float*)d_in[12];
  const float* hW2  = (const float*)d_in[13];
  const float* hb2  = (const float*)d_in[14];
  const float* nW1  = (const float*)d_in[15];
  const float* nb1  = (const float*)d_in[16];
  const float* nW2  = (const float*)d_in[17];
  const float* nb2  = (const float*)d_in[18];
  const float* hn_s = (const float*)d_in[19];
  const float* hn_b = (const float*)d_in[20];
  const float* gW   = (const float*)d_in[21];
  const float* gb   = (const float*)d_in[22];
  float* out = (float*)d_out;
  _Float16* ws = (_Float16*)d_ws;

  // Pre-swizzle weights into B-fragment-major f16 layout in workspace.
  {
    int t;
    t = SZ_E1;  prep_weights<<<(t + 255) / 256, 256, 0, stream>>>(eW1, ws + OFF_E1, 193, 386, E1_NT, t);
    t = SZ_E2;  prep_weights<<<(t + 255) / 256, 256, 0, stream>>>(eW2, ws + OFF_E2, 386, 64,  E2_NT, t);
    t = SZ_H1W; prep_weights<<<(t + 255) / 256, 256, 0, stream>>>(hW1, ws + OFF_H1, 64,  256, H1_NT, t);
    t = SZ_H2W; prep_weights<<<(t + 255) / 256, 256, 0, stream>>>(hW2, ws + OFF_H2, 256, 64,  H2_NT, t);
  }

  (void)hipFuncSetAttribute((const void*)egnn_main,
                            hipFuncAttributeMaxDynamicSharedMemorySize, (int)SM_TOTAL);

  egnn_main<<<BB * NN, 256, SM_TOTAL, stream>>>(
      f0, f1, reldist, nidx, nmask, ln_g, ln_b, eb1, eb2, hb1, hb2,
      nW1, nb1, nW2, nb2, hn_s, hn_b, gW, gb, ws, out);
}
